// HunyuanVideoMidBlock3D_7335804141708
// MI455X (gfx1250) — matrix operand order repaired and checked by reading, not compiled, not run on hardware
//
#include <hip/hip_runtime.h>
#include <hip/hip_bf16.h>
#include <math.h>

// ---------------------------------------------------------------------------
// HunyuanVideo mid-block 3D for MI455X (gfx1250, wave32, WMMA bf16)
//   resnet0 -> block-causal attention -> resnet1
// All matrix math via V_WMMA_F32_16X16X32_BF16 (bf16 in, f32 accumulate).
// Wave tile 32(M) x 64(N): 8 wmma per 32-deep K-step, ~22 FLOP/byte from L2.
// ---------------------------------------------------------------------------

typedef __attribute__((ext_vector_type(16))) __bf16 v16bf;
typedef __attribute__((ext_vector_type(8)))  float  v8f;

#define NCH   512          // channels
#define NPOS  8192         // 8 * 32 * 32
#define NGRP  32
#define CPG   16           // channels per group

// ----- WMMA fragment loaders -----------------------------------------------
// A fragment: 16x32 bf16, source row-major [M][K] (lda in halves).
// Layout: lanes0-15 M=lane, VGPR0-7 K=kb..kb+15 (contiguous, matches B)
//             (kb = k0 for lanes 0-15, k0+16 for lanes 16-31)
__device__ __forceinline__ v16bf load_a_frag(const __bf16* A, int lda,
                                             int m_tile, int k0, int lane) {
    const int m  = m_tile + (lane & 15);
    const int kb = k0 + ((lane & 16) ? 16 : 0);
    const uint32_t* p = reinterpret_cast<const uint32_t*>(A + (size_t)m * lda);
    const int d0 = kb >> 1;
    union { v16bf v; uint32_t u[8]; } r;
#pragma unroll
    for (int j = 0; j < 4; ++j) r.u[j]     = p[d0 + j];        // K = kb..kb+7
#pragma unroll
    for (int j = 0; j < 4; ++j) r.u[4 + j] = p[d0 + 4 + j];    // K = kb+8..kb+15
    return r.v;
}

// B fragment: 32x16 bf16, source is B^T row-major [N][K]; per-lane row ptr.
// Contiguous layout, same as A: lane = N, halves h -> K = k0 + h (+16 for lanes 16-31).
__device__ __forceinline__ v16bf load_b_frag(const __bf16* Brow, int k0, int lane) {
    const int ks = k0 + ((lane & 16) ? 16 : 0);
    const float4* p = reinterpret_cast<const float4*>(Brow + ks);
    union { v16bf v; float4 f[2]; } r;
    r.f[0] = p[0];
    r.f[1] = p[1];
    return r.v;
}

__device__ __forceinline__ v8f wmma_bf16(v16bf a, v16bf b, v8f c) {
    return __builtin_amdgcn_wmma_f32_16x16x32_bf16(false, a, false, b,
                                                   (short)0, c, false, false);
}

// C/D 16x16 f32 tile: VGPR r, lane l -> M = r + (l>=16 ? 8:0), N = l&15.
__device__ __forceinline__ void store_tile(v8f c, int m_t, int n_t, int lane,
                                           void* outp, long ldo, int out_bf16,
                                           const float* biasM, const float* biasN,
                                           const float* res, float scale) {
    const int n    = n_t + (lane & 15);
    const int mofs = (lane & 16) ? 8 : 0;
#pragma unroll
    for (int r = 0; r < 8; ++r) {
        const int m = m_t + mofs + r;
        float v = c[r] * scale;
        if (biasM) v += biasM[m];
        if (biasN) v += biasN[n];
        const size_t idx = (size_t)m * ldo + n;
        if (res) v += res[idx];
        if (out_bf16) ((__bf16*)outp)[idx] = (__bf16)v;
        else          ((float*)outp)[idx]  = v;
    }
}

// ----- Generic GEMM: D[M][N] = scale*(A[M][K] x B^T[N][K]) + biases + res ---
// Block tile 64(M) x 256(N); 8 waves as 2x4, each wave 32x64 (2x4 WMMA frags).
__global__ __launch_bounds__(256)
void gemm_bf16_kernel(const __bf16* __restrict__ A, int lda,
                      const __bf16* __restrict__ B, int ldb, int K,
                      void* __restrict__ outp, long ldo, int out_bf16,
                      const float* __restrict__ biasM,
                      const float* __restrict__ biasN,
                      const float* __restrict__ res, float scale) {
    const int lane = threadIdx.x & 31;
    const int wave = threadIdx.x >> 5;
    const int m_base = blockIdx.y * 64  + (wave >> 2) * 32;
    const int n_base = blockIdx.x * 256 + (wave & 3) * 64;

    const __bf16* brow[4];
#pragma unroll
    for (int j = 0; j < 4; ++j)
        brow[j] = B + (size_t)(n_base + j * 16 + (lane & 15)) * ldb;

    v8f c0[4] = {{}, {}, {}, {}};   // rows m_base .. m_base+15
    v8f c1[4] = {{}, {}, {}, {}};   // rows m_base+16 .. m_base+31

    for (int k0 = 0; k0 < K; k0 += 32) {
        v16bf a0 = load_a_frag(A, lda, m_base,      k0, lane);
        v16bf a1 = load_a_frag(A, lda, m_base + 16, k0, lane);
#pragma unroll
        for (int j = 0; j < 4; ++j) {
            v16bf b = load_b_frag(brow[j], k0, lane);
            c0[j] = wmma_bf16(a0, b, c0[j]);
            c1[j] = wmma_bf16(a1, b, c1[j]);
        }
    }
#pragma unroll
    for (int j = 0; j < 4; ++j) {
        store_tile(c0[j], m_base,      n_base + j * 16, lane, outp, ldo, out_bf16, biasM, biasN, res, scale);
        store_tile(c1[j], m_base + 16, n_base + j * 16, lane, outp, ldo, out_bf16, biasM, biasN, res, scale);
    }
}

// ----- Causal 3x3x3 conv as implicit GEMM ----------------------------------
// act: pos-major bf16 [8192][512]; wpack: bf16 [27][c_out=512][c_in=512]
// out: channel-major f32 [512][8192] (= NCDHW). Replicate pad t(2,0), h/w(1,1).
// Wave n-tile = 64 positions = two image rows (y, y+1) of one frame.
__global__ __launch_bounds__(256)
void conv3d_wmma_kernel(const __bf16* __restrict__ act,
                        const __bf16* __restrict__ wpack,
                        const float* __restrict__ bias,
                        const float* __restrict__ res,
                        float* __restrict__ out) {
    const int lane = threadIdx.x & 31;
    const int wave = threadIdx.x >> 5;
    const int m_base = blockIdx.y * 64  + (wave >> 2) * 32;  // c_out
    const int n_base = blockIdx.x * 256 + (wave & 3) * 64;   // pos (mult of 64)

    const int f  = n_base >> 10;
    const int y  = (n_base >> 5) & 31;   // even; rows y and y+1
    const int x0 = lane & 15;

    v8f c0[4] = {{}, {}, {}, {}};
    v8f c1[4] = {{}, {}, {}, {}};

    for (int t = 0; t < 27; ++t) {
        const int kt = t / 9, ky = (t / 3) % 3, kx = t % 3;
        int fs = f + kt - 2;  if (fs < 0) fs = 0;                 // causal edge pad
        int ya = y + ky - 1;      ya = ya < 0 ? 0 : (ya > 31 ? 31 : ya);
        int yb = y + 1 + ky - 1;  yb = yb < 0 ? 0 : (yb > 31 ? 31 : yb);
        const __bf16* brow[4];
#pragma unroll
        for (int j = 0; j < 4; ++j) {
            const int ys = (j < 2) ? ya : yb;
            int xs = (j & 1) * 16 + x0 + kx - 1;
            xs = xs < 0 ? 0 : (xs > 31 ? 31 : xs);
            brow[j] = act + (size_t)(fs * 1024 + ys * 32 + xs) * NCH;
        }
        const __bf16* W = wpack + (size_t)t * NCH * NCH;
        for (int k0 = 0; k0 < NCH; k0 += 32) {
            v16bf a0 = load_a_frag(W, NCH, m_base,      k0, lane);
            v16bf a1 = load_a_frag(W, NCH, m_base + 16, k0, lane);
#pragma unroll
            for (int j = 0; j < 4; ++j) {
                v16bf b = load_b_frag(brow[j], k0, lane);
                c0[j] = wmma_bf16(a0, b, c0[j]);
                c1[j] = wmma_bf16(a1, b, c1[j]);
            }
        }
    }
#pragma unroll
    for (int j = 0; j < 4; ++j) {
        store_tile(c0[j], m_base,      n_base + j * 16, lane, out, NPOS, 0, bias, nullptr, res, 1.0f);
        store_tile(c1[j], m_base + 16, n_base + j * 16, lane, out, NPOS, 0, bias, nullptr, res, 1.0f);
    }
}

// ----- GroupNorm stats: one block per group, fp64 accumulation -------------
__global__ __launch_bounds__(256)
void gn_stats_kernel(const float* __restrict__ x, float* __restrict__ stats) {
    __shared__ double ssum[256], ssq[256];
    const int g = blockIdx.x, tid = threadIdx.x;
    const float* base = x + (size_t)g * CPG * NPOS;
    double s = 0.0, s2 = 0.0;
    for (int i = tid; i < CPG * NPOS; i += 256) {
        const double v = (double)base[i];
        s += v; s2 += v * v;
    }
    ssum[tid] = s; ssq[tid] = s2;
    __syncthreads();
    for (int st = 128; st > 0; st >>= 1) {
        if (tid < st) { ssum[tid] += ssum[tid + st]; ssq[tid] += ssq[tid + st]; }
        __syncthreads();
    }
    if (tid == 0) {
        const double inv_n = 1.0 / (double)(CPG * NPOS);
        const double mean  = ssum[0] * inv_n;
        const double var   = ssq[0] * inv_n - mean * mean;
        stats[g * 2 + 0] = (float)mean;
        stats[g * 2 + 1] = (float)(1.0 / sqrt(var + 1e-6));
    }
}

// ----- GroupNorm apply (+optional SiLU) with ch-major -> pos-major transpose
__global__ __launch_bounds__(256)
void gn_apply_kernel(const float* __restrict__ x, const float* __restrict__ stats,
                     const float* __restrict__ gamma, const float* __restrict__ beta,
                     __bf16* __restrict__ out, int do_silu) {
    __shared__ float tile[32][33];
    const int p0 = blockIdx.x * 32;          // position base
    const int c0 = blockIdx.y * 32;          // channel base
    const int tp = threadIdx.x & 31;
    const int tr = threadIdx.x >> 5;         // 0..7
#pragma unroll
    for (int it = 0; it < 4; ++it) {
        const int cl = tr + it * 8;
        const int c  = c0 + cl;
        const int g  = c >> 4;
        float v = x[(size_t)c * NPOS + p0 + tp];
        v = (v - stats[g * 2]) * stats[g * 2 + 1] * gamma[c] + beta[c];
        if (do_silu) v = v / (1.0f + __expf(-v));
        tile[cl][tp] = v;
    }
    __syncthreads();
#pragma unroll
    for (int it = 0; it < 4; ++it) {
        const int pl = tr + it * 8;
        out[(size_t)(p0 + pl) * NCH + c0 + tp] = (__bf16)tile[tp][pl];
    }
}

// ----- Row softmax over valid kl columns (block-causal already applied) ----
__global__ __launch_bounds__(256)
void softmax_kernel(const float* __restrict__ S, __bf16* __restrict__ P, int kl) {
    __shared__ float red[256];
    const int i = blockIdx.x, tid = threadIdx.x;
    const float* row = S + (size_t)i * NPOS;
    float m = -1e30f;
    for (int j = tid; j < kl; j += 256) m = fmaxf(m, row[j]);
    red[tid] = m; __syncthreads();
    for (int st = 128; st > 0; st >>= 1) {
        if (tid < st) red[tid] = fmaxf(red[tid], red[tid + st]);
        __syncthreads();
    }
    m = red[0]; __syncthreads();
    float s = 0.0f;
    for (int j = tid; j < kl; j += 256) s += __expf(row[j] - m);
    red[tid] = s; __syncthreads();
    for (int st = 128; st > 0; st >>= 1) {
        if (tid < st) red[tid] += red[tid + st];
        __syncthreads();
    }
    const float inv = 1.0f / red[0];
    __bf16* prow = P + (size_t)i * NPOS;
    for (int j = tid; j < kl; j += 256) prow[j] = (__bf16)(__expf(row[j] - m) * inv);
}

// ----- Weight repacking ----------------------------------------------------
// w (O,I,3,3,3) f32 -> wpack [tap][o][i] bf16
__global__ __launch_bounds__(256)
void repack_conv_kernel(const float* __restrict__ w, __bf16* __restrict__ wp) {
    const int idx = blockIdx.x * 256 + threadIdx.x;   // o*512 + i
    const int o = idx >> 9, i = idx & 511;
    const float* src = w + (size_t)idx * 27;
#pragma unroll
    for (int t = 0; t < 27; ++t)
        wp[((size_t)t * NCH + o) * NCH + i] = (__bf16)src[t];
}

__global__ __launch_bounds__(256)
void cvt_bf16_kernel(const float* __restrict__ src, __bf16* __restrict__ dst, int n) {
    const int i = blockIdx.x * 256 + threadIdx.x;
    if (i < n) dst[i] = (__bf16)src[i];
}

// ---------------------------------------------------------------------------
extern "C" void kernel_launch(void* const* d_in, const int* in_sizes, int n_in,
                              void* d_out, int out_size, void* d_ws, size_t ws_size,
                              hipStream_t stream) {
    (void)in_sizes; (void)n_in; (void)out_size; (void)ws_size;
    const float* x     = (const float*)d_in[0];
    const float* r0_g1 = (const float*)d_in[1];
    const float* r0_b1 = (const float*)d_in[2];
    const float* r0_w1 = (const float*)d_in[3];
    const float* r0_c1 = (const float*)d_in[4];
    const float* r0_g2 = (const float*)d_in[5];
    const float* r0_b2 = (const float*)d_in[6];
    const float* r0_w2 = (const float*)d_in[7];
    const float* r0_c2 = (const float*)d_in[8];
    const float* r1_g1 = (const float*)d_in[9];
    const float* r1_b1 = (const float*)d_in[10];
    const float* r1_w1 = (const float*)d_in[11];
    const float* r1_c1 = (const float*)d_in[12];
    const float* r1_g2 = (const float*)d_in[13];
    const float* r1_b2 = (const float*)d_in[14];
    const float* r1_w2 = (const float*)d_in[15];
    const float* r1_c2 = (const float*)d_in[16];
    const float* a_gng = (const float*)d_in[17];
    const float* a_gnb = (const float*)d_in[18];
    const float* a_wq  = (const float*)d_in[19];
    const float* a_bq  = (const float*)d_in[20];
    const float* a_wk  = (const float*)d_in[21];
    const float* a_bk  = (const float*)d_in[22];
    const float* a_wv  = (const float*)d_in[23];
    const float* a_bv  = (const float*)d_in[24];
    const float* a_wo  = (const float*)d_in[25];
    const float* a_bo  = (const float*)d_in[26];

    // --- workspace layout (256B aligned) ---
    char* ws = (char*)d_ws;
    size_t off = 0;
    auto wsalloc = [&](size_t bytes) -> char* {
        char* p = ws + off;
        off += (bytes + 255) & ~(size_t)255;
        return p;
    };
    float*  stats = (float*) wsalloc(64 * sizeof(float));
    __bf16* ACT   = (__bf16*)wsalloc((size_t)NPOS * NCH * 2);   // pos-major act
    float*  HA    = (float*) wsalloc((size_t)NCH * NPOS * 4);   // ch-major
    float*  HB    = (float*) wsalloc((size_t)NCH * NPOS * 4);   // ch-major
    __bf16* Qb    = (__bf16*)wsalloc((size_t)NPOS * NCH * 2);   // pos-major
    __bf16* Kb    = (__bf16*)wsalloc((size_t)NPOS * NCH * 2);   // pos-major
    __bf16* Vt    = (__bf16*)wsalloc((size_t)NCH * NPOS * 2);   // dim-major
    float*  Sb    = (float*) wsalloc((size_t)1024 * NPOS * 4);  // per-frame scores
    __bf16* Pb    = (__bf16*)wsalloc((size_t)1024 * NPOS * 2);  // per-frame probs
    __bf16* Ob    = (__bf16*)wsalloc((size_t)NPOS * NCH * 2);   // pre-proj attn out
    __bf16* Wc0   = (__bf16*)wsalloc((size_t)27 * NCH * NCH * 2);
    __bf16* Wc1   = (__bf16*)wsalloc((size_t)27 * NCH * NCH * 2);
    __bf16* Wc2   = (__bf16*)wsalloc((size_t)27 * NCH * NCH * 2);
    __bf16* Wc3   = (__bf16*)wsalloc((size_t)27 * NCH * NCH * 2);
    __bf16* Wq    = (__bf16*)wsalloc((size_t)NCH * NCH * 2);
    __bf16* Wk    = (__bf16*)wsalloc((size_t)NCH * NCH * 2);
    __bf16* Wv    = (__bf16*)wsalloc((size_t)NCH * NCH * 2);
    __bf16* Wo    = (__bf16*)wsalloc((size_t)NCH * NCH * 2);

    const dim3 B256(256);

    // --- weight repacks ---
    repack_conv_kernel<<<1024, B256, 0, stream>>>(r0_w1, Wc0);
    repack_conv_kernel<<<1024, B256, 0, stream>>>(r0_w2, Wc1);
    repack_conv_kernel<<<1024, B256, 0, stream>>>(r1_w1, Wc2);
    repack_conv_kernel<<<1024, B256, 0, stream>>>(r1_w2, Wc3);
    cvt_bf16_kernel<<<1024, B256, 0, stream>>>(a_wq, Wq, NCH * NCH);
    cvt_bf16_kernel<<<1024, B256, 0, stream>>>(a_wk, Wk, NCH * NCH);
    cvt_bf16_kernel<<<1024, B256, 0, stream>>>(a_wv, Wv, NCH * NCH);
    cvt_bf16_kernel<<<1024, B256, 0, stream>>>(a_wo, Wo, NCH * NCH);

    auto gn = [&](const float* src, const float* g_, const float* b_, int silu) {
        gn_stats_kernel<<<NGRP, B256, 0, stream>>>(src, stats);
        gn_apply_kernel<<<dim3(NPOS / 32, NCH / 32), B256, 0, stream>>>(src, stats, g_, b_, ACT, silu);
    };
    auto conv = [&](const __bf16* wpk, const float* b_, const float* res_, float* outp) {
        conv3d_wmma_kernel<<<dim3(NPOS / 256, NCH / 64), B256, 0, stream>>>(ACT, wpk, b_, res_, outp);
    };
    auto gemm = [&](const __bf16* A, int lda, const __bf16* Bm, int ldb,
                    int M, int N, int K, void* outp, long ldo, int obf,
                    const float* bM, const float* bN, const float* res_, float scale) {
        gemm_bf16_kernel<<<dim3(N / 256, M / 64), B256, 0, stream>>>(
            A, lda, Bm, ldb, K, outp, ldo, obf, bM, bN, res_, scale);
    };

    // --- resnet block 0 (input x, ch-major) ---
    gn(x, r0_g1, r0_b1, 1);
    conv(Wc0, r0_c1, nullptr, HA);
    gn(HA, r0_g2, r0_b2, 1);
    conv(Wc1, r0_c2, x, HB);              // HB = resnet0 output (ch-major)

    // --- block-causal attention ---
    gn(HB, a_gng, a_gnb, 0);              // ACT = normalized seq (pos-major bf16)
    gemm(ACT, NCH, Wq, NCH, NPOS, NCH, NCH, Qb, NCH, 1, nullptr, a_bq, nullptr, 1.0f);
    gemm(ACT, NCH, Wk, NCH, NPOS, NCH, NCH, Kb, NCH, 1, nullptr, a_bk, nullptr, 1.0f);
    gemm(Wv, NCH, ACT, NCH, NCH, NPOS, NCH, Vt, NPOS, 1, a_bv, nullptr, nullptr, 1.0f);
    const float scl = 0.044194173824159216f;   // 1/sqrt(512)
    for (int f = 0; f < 8; ++f) {
        const int kl = (f + 1) * 1024;         // frame-level causal: keys j < kl
        gemm(Qb + (size_t)f * 1024 * NCH, NCH, Kb, NCH,
             1024, kl, NCH, Sb, NPOS, 0, nullptr, nullptr, nullptr, scl);
        softmax_kernel<<<1024, B256, 0, stream>>>(Sb, Pb, kl);
        gemm(Pb, NPOS, Vt, NPOS, 1024, NCH, kl,
             Ob + (size_t)f * 1024 * NCH, NCH, 1, nullptr, nullptr, nullptr, 1.0f);
    }
    // output projection + residual (in-place on HB, ch-major f32)
    gemm(Wo, NCH, Ob, NCH, NCH, NPOS, NCH, HB, NPOS, 0, a_bo, nullptr, HB, 1.0f);

    // --- resnet block 1 ---
    gn(HB, r1_g1, r1_b1, 1);
    conv(Wc2, r1_c1, nullptr, HA);
    gn(HA, r1_g2, r1_b2, 1);
    conv(Wc3, r1_c2, HB, (float*)d_out);  // final: NCDHW f32 into d_out
}
